// NonLinearTSQ_79551384256577
// MI455X (gfx1250) — matrix operand order repaired and checked by reading, not compiled, not run on hardware
//
#include <hip/hip_runtime.h>
#include <math.h>

typedef __bf16 bf16_t;
typedef __attribute__((ext_vector_type(16))) __bf16 v16bf;
typedef __attribute__((ext_vector_type(8)))  float  v8f;

#define N_SC    64
#define NODES   32768
#define TILE    32
#define THREADS 256
#define DIM_IN  160
#define DIM_OUT 240

// bf16 weight workspace layout (element offsets)
#define WS0_OFF 0                          // Wss0: [u][p][v]    64*64*64
#define WS1_OFF 262144                     // Wsv1: [v*32+q][u]  1024*64
#define WS2_OFF (262144 + 65536)           // Wvv0: [p][u*32+v]  64*1024  (pre-scaled 1/sqrt3)
#define WS3_OFF (262144 + 65536 + 65536)   // Wvv2: [u*16+r][v]  512*32
#define WS_TOTAL 409600

// path normalization: FAN0=64*64+32*32=5120, FAN1=2048, FAN2=1024
__constant__ const float C0 = 0.013975424859373685f;  // 1/sqrt(5120)
__constant__ const float C1 = 0.022097086912079608f;  // 1/sqrt(2048)
__constant__ const float C2 = 0.03125f;               // 1/sqrt(1024)
#define INV_SQRT2 0.7071067811865475f
#define INV_SQRT3 0.5773502691896258f
#define INV_SQRT6 0.4082482904638630f

// ---------------------------------------------------------------------------
// fp32 -> bf16 weight repack (transposed to [N][K] so B-fragment K pairs are
// contiguous per the CDNA5 wave32 B-matrix VGPR layout).
// ---------------------------------------------------------------------------
__global__ void prep_weights(const float* __restrict__ wss0,
                             const float* __restrict__ wsv1,
                             const float* __restrict__ wvv0,
                             const float* __restrict__ wvv2,
                             bf16_t* __restrict__ ws) {
  int idx = blockIdx.x * blockDim.x + threadIdx.x;
  if (idx < 262144) {                         // ss0: dst [u][p][v]
    int u = idx >> 12, rem = idx & 4095, p = rem >> 6, v = rem & 63;
    ws[WS0_OFF + idx] = (bf16_t)wss0[(u * 64 + v) * 64 + p];
  } else if (idx < 262144 + 65536) {          // sv1: dst [v*32+q][u]
    int d = idx - 262144;
    int n = d >> 6, u = d & 63, v = n >> 5, q = n & 31;
    ws[WS1_OFF + d] = (bf16_t)wsv1[(u * 32 + v) * 32 + q];
  } else if (idx < 262144 + 131072) {         // vv0: dst [p][u*32+v], fold 1/sqrt3
    int d = idx - (262144 + 65536);
    int p = d >> 10, k = d & 1023;
    ws[WS2_OFF + d] = (bf16_t)(wvv0[k * 64 + p] * INV_SQRT3);
  } else if (idx < WS_TOTAL) {                // vv2: dst [u*16+r][v]
    int d = idx - (262144 + 131072);
    int n = d >> 5, v = d & 31, u = n >> 4, r = n & 15;
    ws[WS3_OFF + d] = (bf16_t)wvv2[(u * 32 + v) * 16 + r];
  }
}

// ---------------------------------------------------------------------------
// WMMA fragment helpers (wave32 layouts from CDNA5 ISA §7.12.2)
// ---------------------------------------------------------------------------
__device__ __forceinline__ v8f wmma_bf16(v16bf a, v16bf b, v8f c) {
  return __builtin_amdgcn_wmma_f32_16x16x32_bf16(false, a, false, b, (short)0, c,
                                                 false, false);
}

// A (16x32 bf16), source row-major [M][K] in LDS.
// lane l: M = l%16 ; VGPR r: K = kbase + (r>>2)*16 + (r&3)*2 + (l>=16 ? 8 : 0)
__device__ __forceinline__ v16bf load_frag_A(const bf16_t* base, int row0, int ld,
                                             int kbase, int lane) {
  int m  = row0 + (lane & 15);
  int kh = (lane >> 4) << 3;
  const bf16_t* p = base + m * ld + kbase + kh;
  v16bf a;
#pragma unroll
  for (int r = 0; r < 8; ++r) {
    int k = ((r >> 2) << 4) + ((r & 3) << 1);
    a[2 * r]     = p[k];
    a[2 * r + 1] = p[k + 1];
  }
  return a;
}

// B (32x16 bf16), source stored transposed [N][K] (K contiguous).
// lane l: N = l%16 ; VGPR r: K = kbase + (r>>2)*16 + (r&3)*2 + (l>=16 ? 8 : 0)
__device__ __forceinline__ v16bf load_frag_B(const bf16_t* base, int n0, int ldK,
                                             int kbase, int lane) {
  int n  = n0 + (lane & 15);
  int kh = (lane >> 4) << 3;
  const bf16_t* p = base + n * ldK + kbase + kh;
  v16bf b;
#pragma unroll
  for (int r = 0; r < 8; ++r) {
    int k = ((r >> 2) << 4) + ((r & 3) << 1);
    b[2 * r]     = p[k];
    b[2 * r + 1] = p[k + 1];
  }
  return b;
}

// ---------------------------------------------------------------------------
// Main kernel: 32 nodes per block, 8 waves (wave32), bf16 WMMA + fp32 accum.
// LDS ~94 KB -> 3 blocks per WGP (320 KB) for L2-latency hiding.
// ---------------------------------------------------------------------------
__global__ __launch_bounds__(THREADS, 1)
void nltsq_main(const float* __restrict__ feats,
                const float* __restrict__ gates,
                const bf16_t* __restrict__ ws,
                float* __restrict__ out) {
  extern __shared__ char smem[];
  float*  sS  = (float*)(smem);              // [32][64]  tanh(scalars)        8 KB
  bf16_t* sA  = (bf16_t*)(smem + 8192);      // [32][64]  bf16 scalars (A)     4 KB
  float*  sV  = (float*)(smem + 12288);      // [32][32][3] gated vectors     12 KB
  bf16_t* vA  = (bf16_t*)(smem + 24576);     // [96][32]  rows = m*3+j (A)     6 KB
  bf16_t* scr = (bf16_t*)(smem + 30720);     // reused: Gram[32][1024] / U[32][1024] / tv-half[96][256]  64 KB

  const int tid   = threadIdx.x;
  const int lane  = tid & 31;
  const int wv    = tid >> 5;
  const int node0 = blockIdx.x * TILE;

  const bf16_t* Wss0 = ws + WS0_OFF;
  const bf16_t* Wsv1 = ws + WS1_OFF;
  const bf16_t* Wvv0 = ws + WS2_OFF;
  const bf16_t* Wvv2 = ws + WS3_OFF;

  // ---- Phase 0: load tile, apply gate nonlinearity, stage bf16 A operands ----
  for (int t = tid; t < TILE * DIM_IN; t += THREADS) {
    int m = t / DIM_IN, c = t - m * DIM_IN;
    float x = feats[(size_t)(node0 + m) * DIM_IN + c];
    if (c < N_SC) {
      float s = tanhf(x);
      sS[m * 64 + c] = s;
      sA[m * 64 + c] = (bf16_t)s;
    } else {
      int c2 = c - N_SC;
      int vi = c2 / 3, i = c2 - vi * 3;
      float val = x * tanhf(gates[vi]);
      sV[(m * 32 + vi) * 3 + i]  = val;
      vA[(m * 3 + i) * 32 + vi]  = (bf16_t)val;
    }
  }
  __syncthreads();

  // ---- Gram matrix for vv0 path: scr[m][u*32+v] = v_u . v_v ----
  for (int t = tid; t < TILE * 1024; t += THREADS) {
    int m = t >> 10, k = t & 1023;
    int u = k >> 5, v = k & 31;
    const float* pu = &sV[(m * 32 + u) * 3];
    const float* pv = &sV[(m * 32 + v) * 3];
    scr[t] = (bf16_t)(pu[0] * pv[0] + pu[1] * pv[1] + pu[2] * pv[2]);
  }
  __syncthreads();

  // ---- Phase 1+2: out0 = C0 * ( Σ_u s_u (s·W_u)  +  Gram · Wvv0 ) ----
  {
    const int mt = wv >> 2;  // 0..1  (16-node M tile)
    const int pt = wv & 3;   // 0..3  (16-channel N tile of p)
    v8f acc = {0.f, 0.f, 0.f, 0.f, 0.f, 0.f, 0.f, 0.f};
    v16bf aLo = load_frag_A(sA, mt * 16, 64, 0, lane);
    v16bf aHi = load_frag_A(sA, mt * 16, 64, 32, lane);
    for (int u = 0; u < 64; ++u) {
      const bf16_t* Bu = Wss0 + u * 4096;  // [p][v]
      // prefetch this wave's B-slice 2 iterations ahead (32 lanes x 64B = 2 KB)
      if (u < 62) {
        const char* nb = (const char*)(Wss0 + (u + 2) * 4096 + pt * 1024);
        __builtin_prefetch(nb + lane * 64, 0, 1);
      }
      v16bf bLo = load_frag_B(Bu, pt * 16, 64, 0, lane);
      v16bf bHi = load_frag_B(Bu, pt * 16, 64, 32, lane);
      v8f cu = {0.f, 0.f, 0.f, 0.f, 0.f, 0.f, 0.f, 0.f};
      cu = wmma_bf16(aLo, bLo, cu);
      cu = wmma_bf16(aHi, bHi, cu);
#pragma unroll
      for (int r = 0; r < 8; ++r) {        // per-row scale by s[m,u]
        int m = mt * 16 + r + ((lane >> 4) << 3);
        acc[r] += sS[m * 64 + u] * cu[r];
      }
    }
    for (int kt = 0; kt < 32; ++kt) {      // vv0: K = 1024 over Gram
      v16bf a = load_frag_A(scr, mt * 16, 1024, kt * 32, lane);
      v16bf b = load_frag_B(Wvv0, pt * 16, 1024, kt * 32, lane);
      acc = wmma_bf16(a, b, acc);
    }
#pragma unroll
    for (int r = 0; r < 8; ++r) {
      int m = mt * 16 + r + ((lane >> 4) << 3);
      int p = pt * 16 + (lane & 15);
      out[(size_t)(node0 + m) * DIM_OUT + p] = C0 * acc[r];
    }
  }
  __syncthreads();  // Gram reads done; scr free for U

  // ---- Phase 3: U[m][v*32+q] = Σ_u s_u Wsv1[u,v,q]  ([32x64]x[64x1024]) ----
  for (int t = wv; t < 128; t += 8) {
    int mt = t >> 6, nt = t & 63;
    v16bf aLo = load_frag_A(sA, mt * 16, 64, 0, lane);
    v16bf aHi = load_frag_A(sA, mt * 16, 64, 32, lane);
    v16bf bLo = load_frag_B(Wsv1, nt * 16, 64, 0, lane);
    v16bf bHi = load_frag_B(Wsv1, nt * 16, 64, 32, lane);
    v8f c = {0.f, 0.f, 0.f, 0.f, 0.f, 0.f, 0.f, 0.f};
    c = wmma_bf16(aLo, bLo, c);
    c = wmma_bf16(aHi, bHi, c);
#pragma unroll
    for (int r = 0; r < 8; ++r) {
      int m = mt * 16 + r + ((lane >> 4) << 3);
      int n = nt * 16 + (lane & 15);
      scr[m * 1024 + n] = (bf16_t)c[r];
    }
  }
  __syncthreads();
  // epilogue: out1[m,q,i] = C1 * Σ_v U[m,v,q] v[m,v,i]
  for (int t = tid; t < TILE * 96; t += THREADS) {
    int m = t / 96, rem = t - m * 96, q = rem / 3, i = rem - q * 3;
    float a = 0.f;
    for (int v = 0; v < 32; ++v)
      a += (float)scr[m * 1024 + v * 32 + q] * sV[(m * 32 + v) * 3 + i];
    out[(size_t)(node0 + m) * DIM_OUT + 64 + q * 3 + i] = C1 * a;
  }
  __syncthreads();  // U reads done; scr free for tv halves

  // ---- Phase 4: vv2 in two u-halves to cap LDS at 64 KB.
  // half h: tv[(m*3+j)][(u-16h)*16+r] = Σ_v v[m,v,j] Wvv2[u,v,r], u in [16h,16h+16)
  // S accumulators persist in registers across the halves.
  float Sacc[2][9];
#pragma unroll
  for (int pi = 0; pi < 2; ++pi)
#pragma unroll
    for (int e = 0; e < 9; ++e) Sacc[pi][e] = 0.f;

  for (int h = 0; h < 2; ++h) {
    for (int t = wv; t < 96; t += 8) {     // 6 M-tiles x 16 N-tiles
      int mt = t >> 4, nt = t & 15;
      v16bf a = load_frag_A(vA, mt * 16, 32, 0, lane);
      v16bf b = load_frag_B(Wvv2, h * 256 + nt * 16, 32, 0, lane);
      v8f c = {0.f, 0.f, 0.f, 0.f, 0.f, 0.f, 0.f, 0.f};
      c = wmma_bf16(a, b, c);
#pragma unroll
      for (int r = 0; r < 8; ++r) {
        int row = mt * 16 + r + ((lane >> 4) << 3);
        int col = nt * 16 + (lane & 15);
        scr[row * 256 + col] = (bf16_t)c[r];
      }
    }
    __syncthreads();
    // accumulate partial S[i][j] for this u-half
#pragma unroll
    for (int pi = 0; pi < 2; ++pi) {
      int t = tid + pi * THREADS;          // 512 (m,rr) pairs over 256 threads
      int m = t >> 4, rr = t & 15;
      float* S = Sacc[pi];
      for (int uu = 0; uu < 16; ++uu) {
        int u = h * 16 + uu;
        float vx = sV[(m * 32 + u) * 3 + 0];
        float vy = sV[(m * 32 + u) * 3 + 1];
        float vz = sV[(m * 32 + u) * 3 + 2];
        float t0 = (float)scr[(m * 3 + 0) * 256 + uu * 16 + rr];
        float t1 = (float)scr[(m * 3 + 1) * 256 + uu * 16 + rr];
        float t2 = (float)scr[(m * 3 + 2) * 256 + uu * 16 + rr];
        S[0] += vx * t0; S[1] += vx * t1; S[2] += vx * t2;
        S[3] += vy * t0; S[4] += vy * t1; S[5] += vy * t2;
        S[6] += vz * t0; S[7] += vz * t1; S[8] += vz * t2;
      }
    }
    __syncthreads();                       // before next half overwrites scr
  }
  // projection onto the 5 l=2 components
#pragma unroll
  for (int pi = 0; pi < 2; ++pi) {
    int t = tid + pi * THREADS;
    int m = t >> 4, rr = t & 15;
    const float* S = Sacc[pi];             // [i*3+j]
    float* o = &out[(size_t)(node0 + m) * DIM_OUT + 160 + rr * 5];
    o[0] = C2 * INV_SQRT2 * (S[1] + S[3]);                 // xy
    o[1] = C2 * INV_SQRT2 * (S[5] + S[7]);                 // yz
    o[2] = C2 * INV_SQRT6 * (2.f * S[8] - S[0] - S[4]);    // 2z^2-x^2-y^2
    o[3] = C2 * INV_SQRT2 * (S[2] + S[6]);                 // xz
    o[4] = C2 * INV_SQRT2 * (S[0] - S[4]);                 // x^2-y^2
  }
}

extern "C" void kernel_launch(void* const* d_in, const int* in_sizes, int n_in,
                              void* d_out, int out_size, void* d_ws, size_t ws_size,
                              hipStream_t stream) {
  (void)in_sizes; (void)n_in; (void)out_size; (void)ws_size;
  const float* feats = (const float*)d_in[0];
  const float* gates = (const float*)d_in[1];
  const float* wss0  = (const float*)d_in[2];
  const float* wsv1  = (const float*)d_in[3];
  const float* wvv0  = (const float*)d_in[4];
  const float* wvv2  = (const float*)d_in[5];
  bf16_t* ws  = (bf16_t*)d_ws;
  float*  out = (float*)d_out;

  prep_weights<<<(WS_TOTAL + 255) / 256, 256, 0, stream>>>(wss0, wsv1, wvv0, wvv2, ws);

  const size_t lds_bytes = 30720 + 32 * 1024 * sizeof(bf16_t);  // 96256 (~94 KB)
  nltsq_main<<<NODES / TILE, THREADS, lds_bytes, stream>>>(feats, gates, ws, out);
}
